// MultiHeadAttention_40819369181399
// MI455X (gfx1250) — matrix operand order repaired and checked by reading, not compiled, not run on hardware
//
#include <hip/hip_runtime.h>
#include <hip/hip_bf16.h>

// ---------------------------------------------------------------------------
// Multi-head attention forward for MI455X (gfx1250, wave32, WMMA).
//   B=2, S=2048, H=16, d_k=d_v=64, d_model=1024
// All GEMMs use v_wmma_f32_16x16x32_bf16 (bf16 in, f32 accumulate).
// Operands are pre-converted to bf16 and pre-laid-out so every WMMA fragment
// is two contiguous 16-byte loads (global_load_b128 / ds_load_b128).
// ---------------------------------------------------------------------------

#define B_    2
#define S_    2048
#define H_    16
#define DK_   64
#define DM_   1024
#define ROWS_ (B_ * S_)       // 4096 flattened (b,s) rows

typedef __attribute__((ext_vector_type(16))) __bf16          v16bf;
typedef __attribute__((ext_vector_type(8)))  float           v8f;
typedef __attribute__((ext_vector_type(8)))  unsigned short  v8us;

union BfVec {
    v16bf          v;
    unsigned short u[16];
    v8us           h[2];
};

__device__ __forceinline__ unsigned short f2bf(float f) {
    unsigned int x = __float_as_uint(f);
    x += 0x7fffu + ((x >> 16) & 1u);      // round-to-nearest-even
    return (unsigned short)(x >> 16);
}

__device__ __forceinline__ v8f wmma_bf16(const v16bf& a, const v16bf& b, const v8f& c) {
    return __builtin_amdgcn_wmma_f32_16x16x32_bf16(
        false, a, false, b, (short)0, c, false, false);
}

// ---------------------------------------------------------------------------
// Convert f32 -> bf16, straight layout. One v8us store per thread-iter.
// ---------------------------------------------------------------------------
__global__ void cvt_kernel(const float* __restrict__ src,
                           unsigned short* __restrict__ dst, int n8) {
    int j = blockIdx.x * 256 + threadIdx.x;
    if (j >= n8) return;
    const float4* s = (const float4*)src + (size_t)j * 2;
    float4 a = s[0], b = s[1];
    v8us o;
    o[0] = f2bf(a.x); o[1] = f2bf(a.y); o[2] = f2bf(a.z); o[3] = f2bf(a.w);
    o[4] = f2bf(b.x); o[5] = f2bf(b.y); o[6] = f2bf(b.z); o[7] = f2bf(b.w);
    ((v8us*)dst)[j] = o;
}

// ---------------------------------------------------------------------------
// Convert f32 -> bf16 with transpose: dst[slab][n][k] = src[slab][k][n].
// K-major output so WMMA B fragments are contiguous 16B loads.
// ---------------------------------------------------------------------------
__global__ void cvtT_kernel(const float* __restrict__ src,
                            unsigned short* __restrict__ dst, int KD, int ND) {
    const float*    s = src + (size_t)blockIdx.y * KD * ND;
    unsigned short* d = dst + (size_t)blockIdx.y * KD * ND;
    const int total = KD * ND;
    for (int j = blockIdx.x * 256 + threadIdx.x; j < total; j += gridDim.x * 256) {
        int n = j / KD, k = j - n * KD;
        d[j] = f2bf(s[(size_t)k * ND + n]);
    }
}

// ---------------------------------------------------------------------------
// Kernel 1: per-head input projection (bf16 GEMM, no LDS).
//   Xb: (4096 x 1024) bf16, Wt: [H](64 x 1024) bf16 K-major -> Out bf16
//   transposed==0 : Out[h][row][col]   (q, k layout)
//   transposed==1 : Out[h][col][row]   (v layout)
// grid (64 row-blocks, H), block 256 (8 waves); each WG: 64x64 output tile.
// ---------------------------------------------------------------------------
__global__ void proj_kernel(const unsigned short* __restrict__ Xb,
                            const unsigned short* __restrict__ Wt,
                            unsigned short* __restrict__ Out,
                            int transposed) {
    const int h    = blockIdx.y;
    const int rb   = blockIdx.x;
    const int tid  = threadIdx.x;
    const int wave = tid >> 5;
    const int lane = tid & 31;
    const int m    = lane & 15;
    const int half = lane >> 4;
    const int ko   = half * 8;

    const int tm  = wave >> 1;          // 0..3  row tile
    const int tn0 = (wave & 1) * 2;     // 0 or 2 -> two col tiles per wave
    const int arow = rb * 64 + tm * 16 + m;

    const unsigned short* ap0 = Xb + (size_t)arow * DM_;
    const unsigned short* bp0 = Wt + (size_t)h * DK_ * DM_
                                   + (size_t)(tn0 * 16 + m) * DM_;

    v8f acc0 = {};
    v8f acc1 = {};

    for (int kk = 0; kk < DM_ / 32; ++kk) {
        const unsigned short* ap = ap0 + kk * 32;
        const unsigned short* bp = bp0 + kk * 32 + half * 16;
        __builtin_prefetch(ap + 256, 0, 1);
        __builtin_prefetch(bp + 256, 0, 1);

        BfVec A, B0, B1;
        A.h[0]  = *(const v8us*)(ap + half * 16);
        A.h[1]  = *(const v8us*)(ap + half * 16 + 8);
        B0.h[0] = *(const v8us*)(bp);
        B0.h[1] = *(const v8us*)(bp + 8);
        B1.h[0] = *(const v8us*)(bp + 16 * DM_);
        B1.h[1] = *(const v8us*)(bp + 16 * DM_ + 8);

        acc0 = wmma_bf16(A.v, B0.v, acc0);
        acc1 = wmma_bf16(A.v, B1.v, acc1);
    }

#pragma unroll
    for (int r = 0; r < 8; ++r) {
        int row  = rb * 64 + tm * 16 + r + half * 8;
        int col0 = (tn0 + 0) * 16 + m;
        int col1 = (tn0 + 1) * 16 + m;
        if (!transposed) {
            unsigned short* o = Out + (size_t)h * ROWS_ * DK_ + (size_t)row * DK_;
            o[col0] = f2bf(acc0[r]);
            o[col1] = f2bf(acc1[r]);
        } else {
            unsigned short* o = Out + (size_t)h * DK_ * ROWS_;
            o[(size_t)col0 * ROWS_ + row] = f2bf(acc0[r]);
            o[(size_t)col1 * ROWS_ + row] = f2bf(acc1[r]);
        }
    }
}

// ---------------------------------------------------------------------------
// Kernel 2: fused scores -> masked softmax -> attn out -> attn @ V.
// One WG per (query-16-tile, b, h). 8 waves.
// ---------------------------------------------------------------------------
__global__ void attn_kernel(const unsigned short* __restrict__ qb,
                            const unsigned short* __restrict__ kb,
                            const unsigned short* __restrict__ vbT,
                            const unsigned char*  __restrict__ mask,
                            float* __restrict__ attns,
                            unsigned short* __restrict__ ctxb) {
    __shared__ __align__(16) float          sc[16 * S_];     // 128 KB f32 attn
    __shared__ __align__(16) unsigned short sc16[16 * S_];   // 64 KB bf16 attn
    __shared__ float mkf[S_];                                // 8 KB mask 0/1
    __shared__ float partial[4 * 256];                       // 4 KB reduction

    const int qt   = blockIdx.x;
    const int b    = blockIdx.y;
    const int h    = blockIdx.z;
    const int tid  = threadIdx.x;
    const int wave = tid >> 5;
    const int lane = tid & 31;
    const int m    = lane & 15;
    const int half = lane >> 4;
    const int ko   = half * 8;

    for (int i = tid; i < S_; i += 256)
        mkf[i] = (mask[b * S_ + i] != 0) ? 1.0f : 0.0f;
    __syncthreads();

    // ---- scores = (q @ k^T) * 1/sqrt(dk), masked -> LDS -------------------
    const unsigned short* qh = qb + (size_t)h * ROWS_ * DK_
                                  + (size_t)(b * S_ + qt * 16) * DK_;
    BfVec Aq[2];
#pragma unroll
    for (int kh = 0; kh < 2; ++kh) {
        const unsigned short* p = qh + m * DK_ + kh * 32;
        Aq[kh].h[0] = *(const v8us*)(p + half * 16);
        Aq[kh].h[1] = *(const v8us*)(p + half * 16 + 8);
    }

    const unsigned short* kbp = kb + (size_t)h * ROWS_ * DK_ + (size_t)(b * S_) * DK_;
    for (int kt = wave; kt < S_ / 16; kt += 8) {
        const unsigned short* p = kbp + (size_t)(kt * 16 + m) * DK_ + half * 16;
        __builtin_prefetch(p + 8 * 16 * DK_, 0, 1);
        v8f s8 = {};
#pragma unroll
        for (int kh = 0; kh < 2; ++kh) {
            BfVec Bk;
            Bk.h[0] = *(const v8us*)(p + kh * 32);
            Bk.h[1] = *(const v8us*)(p + kh * 32 + 8);
            s8 = wmma_bf16(Aq[kh].v, Bk.v, s8);
        }
        const float mv = mkf[kt * 16 + m];
#pragma unroll
        for (int r = 0; r < 8; ++r) {
            float v = s8[r] * 0.125f;               // 1/sqrt(64)
            v = (mv != 0.0f) ? v : -1e9f;
            sc[(r + half * 8) * S_ + kt * 16 + m] = v;
        }
    }
    __syncthreads();

    // ---- row softmax (2 rows per wave); also emit bf16 copy ----------------
#pragma unroll
    for (int rr = 0; rr < 2; ++rr) {
        const int row = wave * 2 + rr;
        float* rp = sc + row * S_;
        unsigned short* rp16 = sc16 + row * S_;
        float mx = -3.4e38f;
        for (int i = lane; i < S_; i += 32) mx = fmaxf(mx, rp[i]);
#pragma unroll
        for (int off = 16; off > 0; off >>= 1) mx = fmaxf(mx, __shfl_xor(mx, off, 32));
        float sum = 0.0f;
        for (int i = lane; i < S_; i += 32) {
            float e = __expf(rp[i] - mx);
            rp[i] = e;
            sum += e;
        }
#pragma unroll
        for (int off = 16; off > 0; off >>= 1) sum += __shfl_xor(sum, off, 32);
        const float inv = 1.0f / sum;
        for (int i = lane; i < S_; i += 32) {
            float a = rp[i] * inv * mkf[i];
            rp[i]   = a;
            rp16[i] = f2bf(a);
        }
    }
    __syncthreads();

    // ---- write attn block once to d_out (vectorized) -----------------------
    {
        float4* dst = (float4*)(attns + ((size_t)(h * B_ + b) * S_ + qt * 16) * S_);
        const float4* srcv = (const float4*)sc;
        for (int idx = tid; idx < 16 * S_ / 4; idx += 256)
            dst[idx] = srcv[idx];
    }

    // ---- ctx = attn(16x2048) @ v(2048x64) ----------------------------------
    const int vt   = wave & 3;      // N tile (16 cols of dv)
    const int ksel = wave >> 2;     // K half
    v8f c8 = {};
    const unsigned short* vb = vbT + (size_t)h * DK_ * ROWS_ + b * S_
                                   + (size_t)(vt * 16 + m) * ROWS_;
    const unsigned short* a16 = sc16 + m * S_;
    for (int kt2 = ksel * 32; kt2 < ksel * 32 + 32; ++kt2) {
        BfVec Aa, Bv;
        const unsigned short* ap = a16 + kt2 * 32;
        Aa.h[0] = *(const v8us*)(ap + half * 16);
        Aa.h[1] = *(const v8us*)(ap + half * 16 + 8);
        const unsigned short* bp = vb + kt2 * 32 + half * 16;
        Bv.h[0] = *(const v8us*)bp;
        Bv.h[1] = *(const v8us*)(bp + 8);
        c8 = wmma_bf16(Aa.v, Bv.v, c8);
    }

    if (wave >= 4) {
#pragma unroll
        for (int r = 0; r < 8; ++r)
            partial[vt * 256 + (r + half * 8) * 16 + m] = c8[r];
    }
    __syncthreads();
    if (wave < 4) {
#pragma unroll
        for (int r = 0; r < 8; ++r) {
            float v = c8[r] + partial[vt * 256 + (r + half * 8) * 16 + m];
            int q   = qt * 16 + r + half * 8;
            int col = h * DK_ + vt * 16 + m;
            ctxb[(size_t)(b * S_ + q) * DM_ + col] = f2bf(v);
        }
    }
}

// ---------------------------------------------------------------------------
// Kernel 3: out = ctx(4096x1024)bf16 @ Wo(1024x1024, K-major bf16) -> f32
// grid (64 row-blocks, 16 col-blocks), 64x64 tile per WG. No LDS.
// ---------------------------------------------------------------------------
__global__ void outproj_kernel(const unsigned short* __restrict__ ctxb,
                               const unsigned short* __restrict__ Wot,
                               float* __restrict__ out) {
    const int rb   = blockIdx.x;
    const int cb   = blockIdx.y;
    const int tid  = threadIdx.x;
    const int wave = tid >> 5;
    const int lane = tid & 31;
    const int m    = lane & 15;
    const int half = lane >> 4;
    const int ko   = half * 8;
    const int tm   = wave >> 1;
    const int tn0  = (wave & 1) * 2;
    const int arow = rb * 64 + tm * 16 + m;

    const unsigned short* ap0 = ctxb + (size_t)arow * DM_;
    const unsigned short* bp0 = Wot + (size_t)(cb * 64 + tn0 * 16 + m) * DM_;

    v8f acc0 = {};
    v8f acc1 = {};

    for (int kk = 0; kk < DM_ / 32; ++kk) {
        const unsigned short* ap = ap0 + kk * 32;
        const unsigned short* bp = bp0 + kk * 32 + half * 16;
        __builtin_prefetch(ap + 256, 0, 1);
        __builtin_prefetch(bp + 256, 0, 1);

        BfVec A, B0, B1;
        A.h[0]  = *(const v8us*)(ap + half * 16);
        A.h[1]  = *(const v8us*)(ap + half * 16 + 8);
        B0.h[0] = *(const v8us*)(bp);
        B0.h[1] = *(const v8us*)(bp + 8);
        B1.h[0] = *(const v8us*)(bp + 16 * DM_);
        B1.h[1] = *(const v8us*)(bp + 16 * DM_ + 8);

        acc0 = wmma_bf16(A.v, B0.v, acc0);
        acc1 = wmma_bf16(A.v, B1.v, acc1);
    }

#pragma unroll
    for (int r = 0; r < 8; ++r) {
        int row = rb * 64 + tm * 16 + r + half * 8;
        out[(size_t)row * DM_ + cb * 64 + (tn0 + 0) * 16 + m] = acc0[r];
        out[(size_t)row * DM_ + cb * 64 + (tn0 + 1) * 16 + m] = acc1[r];
    }
}

// ---------------------------------------------------------------------------
extern "C" void kernel_launch(void* const* d_in, const int* in_sizes, int n_in,
                              void* d_out, int out_size, void* d_ws, size_t ws_size,
                              hipStream_t stream) {
    const float* Q  = (const float*)d_in[0];
    const float* K  = (const float*)d_in[1];
    const float* V  = (const float*)d_in[2];
    const float* Wq = (const float*)d_in[3];
    const float* Wk = (const float*)d_in[4];
    const float* Wv = (const float*)d_in[5];
    const float* Wo = (const float*)d_in[6];
    const unsigned char* mask = (const unsigned char*)d_in[7];

    float* out   = (float*)d_out;
    float* attns = out + (size_t)ROWS_ * DM_;   // out part first, attns after

    const size_t headElems = (size_t)H_ * ROWS_ * DK_;   // 4.19M
    const size_t xElems    = (size_t)ROWS_ * DM_;        // 4.19M
    const size_t wElems    = (size_t)H_ * DM_ * DK_;     // 1.05M

    unsigned short* p    = (unsigned short*)d_ws;
    unsigned short* qb   = p;  p += headElems;
    unsigned short* kb   = p;  p += headElems;
    unsigned short* vbT  = p;  p += headElems;
    unsigned short* ctxb = p;  p += xElems;
    unsigned short* xqb  = p;  p += xElems;
    unsigned short* xkb  = p;  p += xElems;
    unsigned short* xvb  = p;  p += xElems;
    unsigned short* wqt  = p;  p += wElems;
    unsigned short* wkt  = p;  p += wElems;
    unsigned short* wvt  = p;  p += wElems;
    unsigned short* wot  = p;  p += (size_t)DM_ * DM_;

    // Stage 0: bf16 conversion / re-layout (bandwidth-trivial).
    const int n8 = (int)(xElems / 8);
    cvt_kernel<<<dim3((n8 + 255) / 256), 256, 0, stream>>>(Q, xqb, n8);
    cvt_kernel<<<dim3((n8 + 255) / 256), 256, 0, stream>>>(K, xkb, n8);
    cvt_kernel<<<dim3((n8 + 255) / 256), 256, 0, stream>>>(V, xvb, n8);
    cvtT_kernel<<<dim3(64, H_), 256, 0, stream>>>(Wq, wqt, DM_, DK_);
    cvtT_kernel<<<dim3(64, H_), 256, 0, stream>>>(Wk, wkt, DM_, DK_);
    cvtT_kernel<<<dim3(64, H_), 256, 0, stream>>>(Wv, wvt, DM_, DK_);
    cvtT_kernel<<<dim3(256, 1), 256, 0, stream>>>(Wo, wot, DM_, DM_);

    // Stage 1: per-head projections.
    dim3 gProj(ROWS_ / 64, H_);
    proj_kernel<<<gProj, 256, 0, stream>>>(xqb, wqt, qb, 0);
    proj_kernel<<<gProj, 256, 0, stream>>>(xkb, wkt, kb, 0);
    proj_kernel<<<gProj, 256, 0, stream>>>(xvb, wvt, vbT, 1);

    // Stage 2: fused attention.
    dim3 gAttn(S_ / 16, B_, H_);
    attn_kernel<<<gAttn, 256, 0, stream>>>(qb, kb, vbT, mask, attns, ctxb);

    // Stage 3: output projection.
    dim3 gOut(ROWS_ / 64, DM_ / 64);
    outproj_kernel<<<gOut, 256, 0, stream>>>(ctxb, wot, out);
}
